// Seq2SeqRNN_tf_8701603741866
// MI455X (gfx1250) — matrix operand order repaired and checked by reading, not compiled, not run on hardware
//
#include <hip/hip_runtime.h>
#include <hip/hip_bf16.h>
#include <stdint.h>

// ---------------------------------------------------------------------------
// Types for CDNA5 WMMA (wave32): bf16 A/B fragments, f32 accumulator
// ---------------------------------------------------------------------------
typedef __attribute__((ext_vector_type(16))) __bf16 v16bf;
typedef __attribute__((ext_vector_type(8)))  float  v8f;

union V16BF { uint4 q[2]; v16bf v; };

static __device__ __forceinline__ unsigned short f2bf_rne(float f) {
  unsigned int x = __float_as_uint(f);
  unsigned int r = x + 0x7FFFu + ((x >> 16) & 1u);
  return (unsigned short)(r >> 16);
}

// A fragment: 16x32 (MxK) bf16. Lane L holds row m = mbase + (L&15).
// elements 0..7  -> K = kbase + (L>>4)*8 + e
// elements 8..15 -> K = kbase + 16 + (L>>4)*8 + (e-8)
static __device__ __forceinline__ v16bf load_a_frag(const unsigned short* rowp, int half) {
  V16BF u;
  u.q[0] = *(const uint4*)(rowp + half * 8);
  u.q[1] = *(const uint4*)(rowp + 16 + half * 8);
  return u.v;
}

// B fragment: 32x16 (KxN) bf16 where B[k][n] = W[n0+n][kbase+k] (W row-major [N,K]).
// Lane L holds col n = n0 + (L&15); elements 0..7 -> K = kbase + (L>>4)*8 + e, 8..15 -> K = kbase + 16 + (L>>4)*8 + (e-8)
static __device__ __forceinline__ v16bf load_b_frag(const unsigned short* rowp, int half) {
  V16BF u;
  u.q[0] = *(const uint4*)(rowp + half * 8);
  u.q[1] = *(const uint4*)(rowp + 16 + half * 8);
  return u.v;
}

// ---------------------------------------------------------------------------
// Utility kernels
// ---------------------------------------------------------------------------
__global__ void cvt_bf16_kernel(const float* __restrict__ s, unsigned short* __restrict__ d, int n) {
  int i = blockIdx.x * 256 + threadIdx.x;
  if (i < n) d[i] = f2bf_rne(s[i]);
}

__global__ void zero_u32_kernel(unsigned int* __restrict__ p, int n) {
  int i = blockIdx.x * 256 + threadIdx.x;
  if (i < n) p[i] = 0u;
}

__global__ void init_tok_kernel(int* __restrict__ tok) {
  if (threadIdx.x < 32) tok[threadIdx.x] = 0;  // BOS
}

// xs[s][b][:] = bf16(emb_enc[inp[b][s]][:]);  grid = S*B blocks
__global__ void enc_gather_kernel(const int* __restrict__ inp, const float* __restrict__ emb,
                                  unsigned short* __restrict__ xs) {
  int sb = blockIdx.x;
  int s = sb >> 5, b = sb & 31;
  int t = inp[b * 64 + s];
  const float* erow = emb + (size_t)t * 512;
  unsigned short* orow = xs + (size_t)sb * 512;
  for (int k = threadIdx.x; k < 512; k += 256) orow[k] = f2bf_rne(erow[k]);
}

// e[b][:] = emb_dec_bf16[tok[b]][:];  grid = 32 blocks
__global__ void dec_gather_kernel(const int* __restrict__ tok,
                                  const unsigned short* __restrict__ emb_bf,
                                  unsigned short* __restrict__ e) {
  int b = blockIdx.x;
  const unsigned short* erow = emb_bf + (size_t)tok[b] * 512;
  unsigned short* orow = e + (size_t)b * 512;
  for (int k = threadIdx.x; k < 512; k += 256) orow[k] = erow[k];
}

// ---------------------------------------------------------------------------
// Fused GRU step: h' = (1-z)*n + z*h for a 16-column slice of H per block.
// 6 waves: (src in {x,h}) x (gate in {r,z,n}); each wave computes BOTH 16-row
// M-tiles with two accumulators sharing one B (weight) fragment per K-step:
// halves weight traffic and gives two independent WMMA chains per wave.
// ---------------------------------------------------------------------------
__global__ __launch_bounds__(192) void gru_step_kernel(
    const unsigned short* __restrict__ x_bf, int Kx,
    const unsigned short* __restrict__ h_bf,
    const float* __restrict__ h_f32, int H,
    const unsigned short* __restrict__ Wih,
    const unsigned short* __restrict__ Whh,
    const float* __restrict__ bih, const float* __restrict__ bhh,
    float* __restrict__ hnew_f32, unsigned short* __restrict__ hnew_bf) {
  __shared__ float lds[12 * 256];

  const int tid  = threadIdx.x;
  const int w    = tid >> 5;        // wave id 0..5
  const int lane = tid & 31;
  const int half = lane >> 4;
  const int l16  = lane & 15;
  const int j0   = blockIdx.x * 16; // hidden-column slice

  const int src  = w / 3;           // 0 = x@Wih.T, 1 = h@Whh.T
  const int gate = w % 3;           // 0=r 1=z 2=n

  const unsigned short* A = src ? h_bf : x_bf;
  const unsigned short* W = src ? Whh : Wih;
  const int K = src ? H : Kx;

  const unsigned short* arow0 = A + (size_t)l16 * K;         // batch rows 0-15
  const unsigned short* arow1 = A + (size_t)(16 + l16) * K;  // batch rows 16-31
  const unsigned short* brow  = W + (size_t)(gate * H + j0 + l16) * K;

  v8f c0 = {0.f, 0.f, 0.f, 0.f, 0.f, 0.f, 0.f, 0.f};
  v8f c1 = {0.f, 0.f, 0.f, 0.f, 0.f, 0.f, 0.f, 0.f};
  for (int k = 0; k < K; k += 32) {
    v16bf b  = load_b_frag(brow + k, half);   // shared weight fragment
    v16bf a0 = load_a_frag(arow0 + k, half);
    v16bf a1 = load_a_frag(arow1 + k, half);
    c0 = __builtin_amdgcn_wmma_f32_16x16x32_bf16(false, a0, false, b, (short)0, c0, false, false);
    c1 = __builtin_amdgcn_wmma_f32_16x16x32_bf16(false, a1, false, b, (short)0, c1, false, false);
  }
  // C layout: VGPR v -> row (v + 8*half), col l16. Tile t = src*6 + gate*2 + mtile
  float* tp0 = &lds[(src * 6 + gate * 2 + 0) * 256];
  float* tp1 = &lds[(src * 6 + gate * 2 + 1) * 256];
#pragma unroll
  for (int v = 0; v < 8; ++v) {
    tp0[(v + 8 * half) * 16 + l16] = c0[v];
    tp1[(v + 8 * half) * 16 + l16] = c1[v];
  }
  __syncthreads();

  // elementwise GRU gate math over the [32 x 16] slice
  for (int t = tid; t < 512; t += 192) {
    int b  = t >> 4;
    int jj = t & 15;
    int mt = b >> 4;
    int base = (b & 15) * 16 + jj;
    int j = j0 + jj;
    float gxr = lds[(0 + mt) * 256 + base] + bih[j];
    float gxz = lds[(2 + mt) * 256 + base] + bih[H + j];
    float gxn = lds[(4 + mt) * 256 + base] + bih[2 * H + j];
    float ghr = lds[(6 + mt) * 256 + base] + bhh[j];
    float ghz = lds[(8 + mt) * 256 + base] + bhh[H + j];
    float ghn = lds[(10 + mt) * 256 + base] + bhh[2 * H + j];
    float r = 1.f / (1.f + __expf(-(gxr + ghr)));
    float z = 1.f / (1.f + __expf(-(gxz + ghz)));
    float n = tanhf(gxn + r * ghn);
    float hn = (1.f - z) * n + z * h_f32[(size_t)b * H + j];
    hnew_f32[(size_t)b * H + j] = hn;
    hnew_bf[(size_t)b * H + j] = f2bf_rne(hn);
  }
}

// ---------------------------------------------------------------------------
// C[32,N] = A[32,K] @ W[N,K]^T (+bias). One wave per 16-col tile; the wave
// computes both 16-row M-tiles, sharing one weight fragment per K-step.
// Used for h_enc projection and the per-step logits GEMM (C -> d_out strided).
// ---------------------------------------------------------------------------
__global__ __launch_bounds__(32) void linear_kernel(
    const unsigned short* __restrict__ A,
    const unsigned short* __restrict__ W,
    const float* __restrict__ bias,
    float* __restrict__ C, long long ldc,
    unsigned short* __restrict__ Cb, int ldcb,
    int K) {
  const int lane = threadIdx.x & 31;
  const int half = lane >> 4;
  const int l16  = lane & 15;
  const int n0   = blockIdx.x * 16;

  const unsigned short* arow0 = A + (size_t)l16 * K;
  const unsigned short* arow1 = A + (size_t)(16 + l16) * K;
  const unsigned short* brow  = W + (size_t)(n0 + l16) * K;

  v8f c0 = {0.f, 0.f, 0.f, 0.f, 0.f, 0.f, 0.f, 0.f};
  v8f c1 = {0.f, 0.f, 0.f, 0.f, 0.f, 0.f, 0.f, 0.f};
  for (int k = 0; k < K; k += 32) {
    v16bf b  = load_b_frag(brow + k, half);
    v16bf a0 = load_a_frag(arow0 + k, half);
    v16bf a1 = load_a_frag(arow1 + k, half);
    c0 = __builtin_amdgcn_wmma_f32_16x16x32_bf16(false, a0, false, b, (short)0, c0, false, false);
    c1 = __builtin_amdgcn_wmma_f32_16x16x32_bf16(false, a1, false, b, (short)0, c1, false, false);
  }

  int col = n0 + l16;
  float bv = bias ? bias[col] : 0.f;
#pragma unroll
  for (int v = 0; v < 8; ++v) {
    int m0 = v + 8 * half;
    int m1 = 16 + v + 8 * half;
    float v0 = c0[v] + bv;
    float v1 = c1[v] + bv;
    C[(size_t)m0 * ldc + col] = v0;
    C[(size_t)m1 * ldc + col] = v1;
    if (Cb) {
      Cb[(size_t)m0 * ldcb + col] = f2bf_rne(v0);
      Cb[(size_t)m1 * ldcb + col] = f2bf_rne(v1);
    }
  }
}

// ---------------------------------------------------------------------------
// Row argmax over V=32000, first-max tie rule (matches jnp.argmax). 1 block/row
// ---------------------------------------------------------------------------
__global__ __launch_bounds__(256) void argmax_kernel(
    const float* __restrict__ logits, long long row_stride, int V, int* __restrict__ tok) {
  __shared__ float sv[256];
  __shared__ int   si[256];
  int b = blockIdx.x;
  const float* row = logits + (size_t)b * row_stride;
  float best = -__builtin_inff();
  int bi = 0;
  for (int v = threadIdx.x; v < V; v += 256) {
    float f = row[v];
    if (f > best) { best = f; bi = v; }
  }
  sv[threadIdx.x] = best;
  si[threadIdx.x] = bi;
  __syncthreads();
  for (int s = 128; s > 0; s >>= 1) {
    if (threadIdx.x < s) {
      float ov = sv[threadIdx.x + s];
      int   oi = si[threadIdx.x + s];
      if (ov > sv[threadIdx.x] || (ov == sv[threadIdx.x] && oi < si[threadIdx.x])) {
        sv[threadIdx.x] = ov;
        si[threadIdx.x] = oi;
      }
    }
    __syncthreads();
  }
  if (threadIdx.x == 0) tok[b] = si[0];
}

// ---------------------------------------------------------------------------
// Host orchestration
// ---------------------------------------------------------------------------
extern "C" void kernel_launch(void* const* d_in, const int* in_sizes, int n_in,
                              void* d_out, int out_size, void* d_ws, size_t ws_size,
                              hipStream_t stream) {
  (void)in_sizes; (void)n_in; (void)out_size; (void)ws_size;
  constexpr int B = 32, S = 64, V = 32000, E = 512, H = 1024, HD = 512, MAX_OUT = 64;

  const int*   inp       = (const int*)  d_in[0];
  const float* emb_enc   = (const float*)d_in[1];
  const float* enc_Wih0  = (const float*)d_in[2];
  const float* enc_Whh0  = (const float*)d_in[3];
  const float* enc_bih0  = (const float*)d_in[4];
  const float* enc_bhh0  = (const float*)d_in[5];
  const float* enc_Wih1  = (const float*)d_in[6];
  const float* enc_Whh1  = (const float*)d_in[7];
  const float* enc_bih1  = (const float*)d_in[8];
  const float* enc_bhh1  = (const float*)d_in[9];
  const float* out_enc_W = (const float*)d_in[10];
  const float* emb_dec   = (const float*)d_in[11];
  const float* dec_Wih0  = (const float*)d_in[12];
  const float* dec_Whh0  = (const float*)d_in[13];
  const float* dec_bih0  = (const float*)d_in[14];
  const float* dec_bhh0  = (const float*)d_in[15];
  const float* dec_Wih1  = (const float*)d_in[16];
  const float* dec_Whh1  = (const float*)d_in[17];
  const float* dec_bih1  = (const float*)d_in[18];
  const float* dec_bhh1  = (const float*)d_in[19];
  const float* out_b     = (const float*)d_in[20];
  float* out = (float*)d_out;

  // --- workspace carve-up (256B aligned) ---
  char* ws = (char*)d_ws;
  size_t off = 0;
  auto carve = [&](size_t bytes) -> void* {
    void* p = ws + off;
    off = (off + bytes + 255) & ~(size_t)255;
    return p;
  };
  unsigned short* wWih0   = (unsigned short*)carve((size_t)3 * H * E  * 2);
  unsigned short* wWhh0   = (unsigned short*)carve((size_t)3 * H * H  * 2);
  unsigned short* wWih1   = (unsigned short*)carve((size_t)3 * H * H  * 2);
  unsigned short* wWhh1   = (unsigned short*)carve((size_t)3 * H * H  * 2);
  unsigned short* wOutEnc = (unsigned short*)carve((size_t)E * H      * 2);
  unsigned short* wEmbDec = (unsigned short*)carve((size_t)V * E      * 2);
  unsigned short* wDih0   = (unsigned short*)carve((size_t)3 * HD * E * 2);
  unsigned short* wDhh0   = (unsigned short*)carve((size_t)3 * HD * HD * 2);
  unsigned short* wDih1   = (unsigned short*)carve((size_t)3 * HD * HD * 2);
  unsigned short* wDhh1   = (unsigned short*)carve((size_t)3 * HD * HD * 2);
  unsigned short* xs0     = (unsigned short*)carve((size_t)S * B * E * 2);
  unsigned short* ys0     = (unsigned short*)carve((size_t)S * B * H * 2);
  unsigned short* zhbf    = (unsigned short*)carve((size_t)B * H * 2);
  float* e0hf[2] = {(float*)carve((size_t)B * H * 4), (float*)carve((size_t)B * H * 4)};
  float* e1hf[2] = {(float*)carve((size_t)B * H * 4), (float*)carve((size_t)B * H * 4)};
  unsigned short* e1hb[2] = {(unsigned short*)carve((size_t)B * H * 2),
                             (unsigned short*)carve((size_t)B * H * 2)};
  float* hd1f[2] = {(float*)carve((size_t)B * HD * 4), (float*)carve((size_t)B * HD * 4)};
  unsigned short* hd1b[2] = {(unsigned short*)carve((size_t)B * HD * 2),
                             (unsigned short*)carve((size_t)B * HD * 2)};
  float* hd2f[2] = {(float*)carve((size_t)B * HD * 4), (float*)carve((size_t)B * HD * 4)};
  unsigned short* hd2b[2] = {(unsigned short*)carve((size_t)B * HD * 2),
                             (unsigned short*)carve((size_t)B * HD * 2)};
  unsigned short* e_emb = (unsigned short*)carve((size_t)B * E * 2);
  int* tok = (int*)carve(32 * 4);

  auto cvt = [&](const float* s, unsigned short* d, int n) {
    cvt_bf16_kernel<<<(n + 255) / 256, 256, 0, stream>>>(s, d, n);
  };
  // --- weight conversion f32 -> bf16 ---
  cvt(enc_Wih0,  wWih0,   3 * H * E);
  cvt(enc_Whh0,  wWhh0,   3 * H * H);
  cvt(enc_Wih1,  wWih1,   3 * H * H);
  cvt(enc_Whh1,  wWhh1,   3 * H * H);
  cvt(out_enc_W, wOutEnc, E * H);
  cvt(emb_dec,   wEmbDec, V * E);
  cvt(dec_Wih0,  wDih0,   3 * HD * E);
  cvt(dec_Whh0,  wDhh0,   3 * HD * HD);
  cvt(dec_Wih1,  wDih1,   3 * HD * HD);
  cvt(dec_Whh1,  wDhh1,   3 * HD * HD);

  // --- zero initial hidden states ---
  zero_u32_kernel<<<(B * H + 255) / 256, 256, 0, stream>>>((unsigned int*)e0hf[0], B * H);
  zero_u32_kernel<<<(B * H + 255) / 256, 256, 0, stream>>>((unsigned int*)e1hf[0], B * H);
  zero_u32_kernel<<<(B * H / 2 + 255) / 256, 256, 0, stream>>>((unsigned int*)e1hb[0], B * H / 2);
  zero_u32_kernel<<<(B * H / 2 + 255) / 256, 256, 0, stream>>>((unsigned int*)zhbf, B * H / 2);

  // --- encoder embedding gather (f32 -> bf16) ---
  enc_gather_kernel<<<S * B, 256, 0, stream>>>(inp, emb_enc, xs0);

  // --- encoder layer 0: bf16 h lives in ys0 (doubles as layer-1 input) ---
  for (int s = 0; s < S; ++s) {
    const unsigned short* hbf = (s == 0) ? zhbf : ys0 + (size_t)(s - 1) * B * H;
    gru_step_kernel<<<H / 16, 192, 0, stream>>>(
        xs0 + (size_t)s * B * E, E, hbf, e0hf[s & 1], H,
        wWih0, wWhh0, enc_bih0, enc_bhh0,
        e0hf[(s + 1) & 1], ys0 + (size_t)s * B * H);
  }
  // --- encoder layer 1 ---
  for (int s = 0; s < S; ++s) {
    gru_step_kernel<<<H / 16, 192, 0, stream>>>(
        ys0 + (size_t)s * B * H, H, e1hb[s & 1], e1hf[s & 1], H,
        wWih1, wWhh1, enc_bih1, enc_bhh1,
        e1hf[(s + 1) & 1], e1hb[(s + 1) & 1]);
  }

  // --- h_enc = [h1T; h2T] @ out_enc_W.T -> decoder initial states ---
  const unsigned short* h1T = ys0 + (size_t)(S - 1) * B * H;  // layer0 final (bf16)
  const unsigned short* h2T = e1hb[0];                        // layer1 final (64 steps -> idx 0)
  linear_kernel<<<HD / 16, 32, 0, stream>>>(h1T, wOutEnc, nullptr,
                                            hd1f[0], HD, hd1b[0], HD, H);
  linear_kernel<<<HD / 16, 32, 0, stream>>>(h2T, wOutEnc, nullptr,
                                            hd2f[0], HD, hd2b[0], HD, H);

  // --- greedy decoder ---
  init_tok_kernel<<<1, 32, 0, stream>>>(tok);
  for (int it = 0; it < MAX_OUT; ++it) {
    dec_gather_kernel<<<B, 256, 0, stream>>>(tok, wEmbDec, e_emb);
    gru_step_kernel<<<HD / 16, 192, 0, stream>>>(
        e_emb, E, hd1b[it & 1], hd1f[it & 1], HD,
        wDih0, wDhh0, dec_bih0, dec_bhh0,
        hd1f[(it + 1) & 1], hd1b[(it + 1) & 1]);
    gru_step_kernel<<<HD / 16, 192, 0, stream>>>(
        hd1b[(it + 1) & 1], HD, hd2b[it & 1], hd2f[it & 1], HD,
        wDih1, wDhh1, dec_bih1, dec_bhh1,
        hd2f[(it + 1) & 1], hd2b[(it + 1) & 1]);
    // logits[b, it, :] = hd2 @ emb_dec.T + out_b  (writes straight into d_out)
    linear_kernel<<<V / 16, 32, 0, stream>>>(
        hd2b[(it + 1) & 1], wEmbDec, out_b,
        out + (size_t)it * V, (long long)MAX_OUT * V, nullptr, 0, HD);
    argmax_kernel<<<B, 256, 0, stream>>>(out + (size_t)it * V, (long long)MAX_OUT * V, V, tok);
  }
}
